// Qwen3MoeSparseMoeBlock_44796508897337
// MI455X (gfx1250) — matrix operand order repaired and checked by reading, not compiled, not run on hardware
//
#include <hip/hip_runtime.h>
#include <math.h>

// ---------------- problem constants (from reference) ----------------
#define TOKENS   4096          // B*S = 4*1024
#define DDIM     2048          // hidden dim
#define MDIM     768           // expert intermediate dim
#define NEXP     32            // num experts
#define KTOP     8             // top-k
#define TKROWS   (TOKENS*KTOP) // 32768 expert-rows

typedef __attribute__((ext_vector_type(16))) __bf16 v16bf;
typedef __attribute__((ext_vector_type(8)))  float  v8f;
typedef unsigned int  v4u __attribute__((ext_vector_type(4)));
typedef int           v4i __attribute__((ext_vector_type(4)));
typedef int           v8i __attribute__((ext_vector_type(8)));
typedef unsigned long long ull;

union BF16Frag { v16bf v; unsigned u[8]; ull d[4]; };
union F32Acc   { v8f v; float f[8]; };

__device__ __forceinline__ unsigned short bfbits(float f) {
  union { __bf16 h; unsigned short s; } c; c.h = (__bf16)f; return c.s;
}
__device__ __forceinline__ unsigned pack2(float a, float b) {
  return (unsigned)bfbits(a) | ((unsigned)bfbits(b) << 16);
}
__device__ __forceinline__ float bf2f_lo(unsigned u) {
  union { unsigned u; float f; } c; c.u = u << 16; return c.f;
}
__device__ __forceinline__ float bf2f_hi(unsigned u) {
  union { unsigned u; float f; } c; c.u = u & 0xFFFF0000u; return c.f;
}

// ---------------- Tensor Data Mover: 2D bf16 tile -> padded LDS rows ----------------
// Copies `rows` x 32 bf16 (64B) from a row-major bf16 tensor (row stride `gstride`
// elems) into LDS rows padded to 72B (pad_interval=16 dwords, pad_amount=2 dwords),
// matching the [*][36]-short LDS layout used by the WMMA fragment loaders.
__device__ __forceinline__ void tdm_load_tile_bf16(const unsigned short* gptr,
                                                   unsigned ldsAddr, int rows,
                                                   unsigned gstride) {
  ull ga = (ull)(uintptr_t)gptr;
  v4u g0;
  g0[0] = 1u;                                        // count=1 (user mode)
  g0[1] = ldsAddr;                                   // LDS byte address
  g0[2] = (unsigned)(ga & 0xFFFFFFFFu);
  g0[3] = (unsigned)((ga >> 32) & 0x01FFFFFFu) | (2u << 30);  // type=2 (image)
  v8i g1;
  g1[0] = (int)((1u << 16) |      // data_size = 2 bytes
                (1u << 20) |      // pad_enable
                (3u << 22) |      // pad_interval: 16 dwords (64B) between pads
                (1u << 25));      // pad_amount: 2 dwords (8B)
  g1[1] = (int)((gstride & 0xFFFFu) << 16);                  // tensor_dim0[15:0]
  g1[2] = (int)((gstride >> 16) | (((unsigned)(TKROWS + 128) & 0xFFFFu) << 16));
  g1[3] = (int)((((unsigned)(TKROWS + 128)) >> 16) | (32u << 16)); // tile_dim0=32
  g1[4] = rows;                                              // tile_dim1, tile_dim2=0
  g1[5] = (int)gstride;                                      // tensor_dim0_stride lo32
  g1[6] = 0;
  g1[7] = 0;
  v4i z = {0, 0, 0, 0};
#if __has_builtin(__builtin_amdgcn_tensor_load_to_lds)
#if __clang_major__ >= 23
  v8i z8 = {0, 0, 0, 0, 0, 0, 0, 0};
  __builtin_amdgcn_tensor_load_to_lds(g0, g1, z, z, z8, 0);
#else
  __builtin_amdgcn_tensor_load_to_lds(g0, g1, z, z, 0);
#endif
#else
  (void)g0; (void)g1; (void)z;
#endif
}
__device__ __forceinline__ void tdm_wait() {
#if __has_builtin(__builtin_amdgcn_s_wait_tensorcnt)
  __builtin_amdgcn_s_wait_tensorcnt(0);
#endif
}

// ---------------- 0. zero histogram state ----------------
__global__ void init_kernel(int* __restrict__ counts, int* __restrict__ cursors) {
  int i = threadIdx.x;
  if (i < NEXP) { counts[i] = 0; cursors[i] = 0; }
}

// ---------------- 1. router: softmax + top-8 + renorm (wave32 lane == expert) ----------------
__global__ __launch_bounds__(256) void router_kernel(
    const float* __restrict__ x, const float* __restrict__ gk,
    float* __restrict__ rw, int* __restrict__ sel, int* __restrict__ counts)
{
  int lane = threadIdx.x & 31;
  int wave = threadIdx.x >> 5;
  int t = blockIdx.x * 8 + wave;
  if (t >= TOKENS) return;
  const float* xr = x + (size_t)t * DDIM;

  float acc = 0.f;
  for (int d = 0; d < DDIM; d += 4) {
    float4 xv = *(const float4*)(xr + d);
    acc += xv.x * gk[(d+0)*NEXP + lane];
    acc += xv.y * gk[(d+1)*NEXP + lane];
    acc += xv.z * gk[(d+2)*NEXP + lane];
    acc += xv.w * gk[(d+3)*NEXP + lane];
  }
  float m = acc;
  for (int off = 16; off > 0; off >>= 1) m = fmaxf(m, __shfl_xor(m, off));
  float p = __expf(acc - m);
  float s = p;
  for (int off = 16; off > 0; off >>= 1) s += __shfl_xor(s, off);
  float prob = p / s;

  float topv[KTOP]; int topi[KTOP];
  float cur = prob, sum8 = 0.f;
#pragma unroll
  for (int k = 0; k < KTOP; ++k) {
    float v = cur; int i = lane;
    for (int off = 16; off > 0; off >>= 1) {
      float ov = __shfl_xor(v, off);
      int   oi = __shfl_xor(i, off);
      if (ov > v || (ov == v && oi < i)) { v = ov; i = oi; }
    }
    topv[k] = v; topi[k] = i; sum8 += v;
    if (lane == i) cur = -1.f;
  }
  if (lane == 0) {
    float inv = __builtin_amdgcn_rcpf(sum8);
#pragma unroll
    for (int k = 0; k < KTOP; ++k) {
      rw[t*KTOP + k]  = topv[k] * inv;
      sel[t*KTOP + k] = topi[k];
      atomicAdd(&counts[topi[k]], 1);
    }
  }
}

// ---------------- 2. exclusive scan over 32 counts ----------------
__global__ void scan_kernel(const int* __restrict__ counts, int* __restrict__ offsets) {
  if (threadIdx.x == 0) {
    int a = 0;
    for (int e = 0; e < NEXP; ++e) { offsets[e] = a; a += counts[e]; }
    offsets[NEXP] = a;
  }
}

// ---------------- 3. scatter token-expert pairs into sorted order ----------------
__global__ void assign_kernel(const int* __restrict__ sel, const int* __restrict__ offsets,
                              int* __restrict__ cursors, int* __restrict__ pos_of,
                              int* __restrict__ row_tok)
{
  int idx = blockIdx.x * blockDim.x + threadIdx.x;
  if (idx >= TKROWS) return;
  int e = sel[idx];
  int pos = offsets[e] + atomicAdd(&cursors[e], 1);
  pos_of[idx] = pos;
  row_tok[pos] = idx / KTOP;
}

// ---------------- WMMA bf16 fragment helpers (wave32 ISA layouts) ----------------
// A (16x32): lanes 0-15 row=lane, K={0..7,16..23}; lanes 16-31 row=lane-16, K={8..15,24..31}
// B (32x16): lanes 0-15 col=lane, K={0..7,16..23}; lanes 16-31 col=lane-16, K={8..15,24..31}
// C (16x16): lanes 0-15 N=lane M=vgpr; lanes 16-31 N=lane-16 M=8+vgpr
#define LDS_STR 36   // row stride in shorts (72B): 8B-aligned b64 frag loads

__device__ __forceinline__ void load_a_frag(BF16Frag& f, const unsigned short* row, int lane) {
  int kh = (lane >> 4) * 8;
  f.d[0] = *(const ull*)(row + kh);
  f.d[1] = *(const ull*)(row + kh + 4);
  f.d[2] = *(const ull*)(row + 16 + kh);
  f.d[3] = *(const ull*)(row + 16 + kh + 4);
}
__device__ __forceinline__ void load_b_frag(BF16Frag& f, const unsigned short* col, int lane) {
  int kh = (lane >> 4) * 8;
  f.d[0] = *(const ull*)(col + kh);
  f.d[1] = *(const ull*)(col + kh + 4);
  f.d[2] = *(const ull*)(col + 16 + kh);
  f.d[3] = *(const ull*)(col + 16 + kh + 4);
}

// ---------------- 4. grouped GEMM #1: inter = silu(x Wg) * (x Wu)  [bf16 out] ----------------
// 128x64 tile, K-step 32, 8 waves as 4(M) x 2(N); double-buffered LDS, reg-staged pipeline
#define G1_MT 128
#define G1_NT 64
__global__ __launch_bounds__(256) void gemm1_kernel(
    const float* __restrict__ x, const float* __restrict__ wg, const float* __restrict__ wu,
    const int* __restrict__ row_tok, const int* __restrict__ offsets,
    unsigned short* __restrict__ interbf)
{
  __shared__ unsigned short sA [2][G1_MT][LDS_STR];
  __shared__ unsigned short sBg[2][G1_NT][LDS_STR];
  __shared__ unsigned short sBu[2][G1_NT][LDS_STR];

  int e = blockIdx.z;
  int rowStart = offsets[e];
  int cnt = offsets[e+1] - rowStart;
  int mBase = blockIdx.x * G1_MT;
  if (mBase >= cnt) return;
  int nBase = blockIdx.y * G1_NT;

  const float* wge = wg + (size_t)e * DDIM * MDIM;
  const float* wue = wu + (size_t)e * DDIM * MDIM;

  int tid = threadIdx.x, lane = tid & 31, wave = tid >> 5;
  int waveM = wave & 3, waveN = wave >> 2;

  // hoisted staging geometry (token gather done once; pointers bumped each K-step)
  const float* aP[4]; int rA[4], kA[4];
#pragma unroll
  for (int it = 0; it < 4; ++it) {
    int l = tid + 256 * it;
    rA[it] = l >> 3; kA[it] = (l & 7) << 2;
    int gRow = mBase + rA[it];
    int src = rowStart + (gRow < cnt ? gRow : cnt - 1);
    aP[it] = x + (size_t)row_tok[src] * DDIM + kA[it];
  }
  const float* gP[2]; const float* uP[2]; int kB[2], nB[2];
#pragma unroll
  for (int it = 0; it < 2; ++it) {
    int l = tid + 256 * it;
    kB[it] = l >> 4; nB[it] = (l & 15) << 2;
    size_t go = (size_t)kB[it] * MDIM + nBase + nB[it];
    gP[it] = wge + go; uP[it] = wue + go;
  }

  F32Acc accG[2][2], accU[2][2];
#pragma unroll
  for (int i = 0; i < 2; ++i)
#pragma unroll
    for (int j = 0; j < 2; ++j) { accG[i][j].v = (v8f)(0.f); accU[i][j].v = (v8f)(0.f); }

  float4 aR[4], gR[2], uR[2];
  auto loadRegs = [&]() {
#pragma unroll
    for (int it = 0; it < 4; ++it) { aR[it] = *(const float4*)aP[it]; aP[it] += 32; }
#pragma unroll
    for (int it = 0; it < 2; ++it) {
      gR[it] = *(const float4*)gP[it]; gP[it] += 32 * MDIM;
      uR[it] = *(const float4*)uP[it]; uP[it] += 32 * MDIM;
    }
  };
  auto storeRegs = [&](int buf) {
#pragma unroll
    for (int it = 0; it < 4; ++it) {
      uint2 pk; pk.x = pack2(aR[it].x, aR[it].y); pk.y = pack2(aR[it].z, aR[it].w);
      *(uint2*)&sA[buf][rA[it]][kA[it]] = pk;
    }
#pragma unroll
    for (int it = 0; it < 2; ++it) {
      int k = kB[it], n = nB[it];
      sBg[buf][n+0][k] = bfbits(gR[it].x); sBg[buf][n+1][k] = bfbits(gR[it].y);
      sBg[buf][n+2][k] = bfbits(gR[it].z); sBg[buf][n+3][k] = bfbits(gR[it].w);
      sBu[buf][n+0][k] = bfbits(uR[it].x); sBu[buf][n+1][k] = bfbits(uR[it].y);
      sBu[buf][n+2][k] = bfbits(uR[it].z); sBu[buf][n+3][k] = bfbits(uR[it].w);
    }
  };

  loadRegs();
  storeRegs(0);
  __syncthreads();

  const int KSTEPS = DDIM / 32;  // 64
  int rowA = waveM * 32 + (lane & 15);
  int colB = waveN * 32 + (lane & 15);
#pragma unroll 2
  for (int kbi = 0; kbi < KSTEPS; ++kbi) {
    int cur = kbi & 1, nxt = cur ^ 1;
    bool more = (kbi + 1) < KSTEPS;
    if (more) loadRegs();

    BF16Frag aF[2], gF[2], uF[2];
#pragma unroll
    for (int si = 0; si < 2; ++si) load_a_frag(aF[si], &sA[cur][rowA + si*16][0], lane);
#pragma unroll
    for (int sj = 0; sj < 2; ++sj) {
      load_b_frag(gF[sj], &sBg[cur][colB + sj*16][0], lane);
      load_b_frag(uF[sj], &sBu[cur][colB + sj*16][0], lane);
    }
#pragma unroll
    for (int si = 0; si < 2; ++si)
#pragma unroll
      for (int sj = 0; sj < 2; ++sj) {
        accG[si][sj].v = __builtin_amdgcn_wmma_f32_16x16x32_bf16(
            false, aF[si].v, false, gF[sj].v, (short)0, accG[si][sj].v, false, false);
        accU[si][sj].v = __builtin_amdgcn_wmma_f32_16x16x32_bf16(
            false, aF[si].v, false, uF[sj].v, (short)0, accU[si][sj].v, false, false);
      }
    if (more) storeRegs(nxt);
    __syncthreads();
  }

  // epilogue: inter = silu(g) * u, stored as bf16
  int rowOff = (lane >> 4) * 8, nOff = lane & 15;
#pragma unroll
  for (int si = 0; si < 2; ++si)
#pragma unroll
    for (int sj = 0; sj < 2; ++sj)
#pragma unroll
      for (int i = 0; i < 8; ++i) {
        int gRow = mBase + waveM * 32 + si * 16 + rowOff + i;
        if (gRow < cnt) {
          float g = accG[si][sj].f[i];
          float u = accU[si][sj].f[i];
          float sig = __builtin_amdgcn_rcpf(1.f + __expf(-g));
          interbf[(size_t)(rowStart + gRow) * MDIM + nBase + waveN*32 + sj*16 + nOff] =
              bfbits(g * sig * u);
        }
      }
}

// ---------------- 5. grouped GEMM #2: eout = inter @ Wd  [A via TDM, bf16 out] ----------------
// 128x128 tile, 8 waves as 2(M) x 4(N); A tiles DMA'd by the Tensor Data Mover into
// padded LDS rows, double-buffered on TENSORcnt; B staged manually (fp32 -> bf16).
#define G2_MT 128
#define G2_NT 128
__global__ __launch_bounds__(256) void gemm2_kernel(
    const unsigned short* __restrict__ interbf, const float* __restrict__ wd,
    const int* __restrict__ offsets, unsigned short* __restrict__ eout)
{
  __shared__ unsigned short sA[2][G2_MT][LDS_STR];
  __shared__ unsigned short sB[2][G2_NT][LDS_STR];

  int e = blockIdx.z;
  int rowStart = offsets[e];
  int cnt = offsets[e+1] - rowStart;
  int mBase = blockIdx.x * G2_MT;
  if (mBase >= cnt) return;
  int nBase = blockIdx.y * G2_NT;

  const float* wde = wd + (size_t)e * MDIM * DDIM;
  const unsigned short* aCur = interbf + (size_t)(rowStart + mBase) * MDIM;

  int tid = threadIdx.x, lane = tid & 31, wave = tid >> 5;
  int waveM = wave & 1, waveN = wave >> 1;

  const float* bP[4]; int kB[4], nB[4];
#pragma unroll
  for (int it = 0; it < 4; ++it) {
    int l = tid + 256 * it;
    kB[it] = l >> 5; nB[it] = (l & 31) << 2;
    bP[it] = wde + (size_t)kB[it] * DDIM + nBase + nB[it];
  }

  F32Acc acc[4][2];
#pragma unroll
  for (int i = 0; i < 4; ++i)
#pragma unroll
    for (int j = 0; j < 2; ++j) acc[i][j].v = (v8f)(0.f);

  float4 bR[4];
  auto loadB = [&]() {
#pragma unroll
    for (int it = 0; it < 4; ++it) { bR[it] = *(const float4*)bP[it]; bP[it] += 32 * DDIM; }
  };
  auto storeB = [&](int buf) {
#pragma unroll
    for (int it = 0; it < 4; ++it) {
      int k = kB[it], n = nB[it];
      sB[buf][n+0][k] = bfbits(bR[it].x); sB[buf][n+1][k] = bfbits(bR[it].y);
      sB[buf][n+2][k] = bfbits(bR[it].z); sB[buf][n+3][k] = bfbits(bR[it].w);
    }
  };
  auto issueA = [&](int buf) {   // TDM, wave 0 only (EXEC ignored by TDM); caller bumps aCur
    tdm_load_tile_bf16(aCur, (unsigned)(uintptr_t)&sA[buf][0][0], G2_MT, MDIM);
  };

  if (wave == 0) issueA(0);
  aCur += 32;
  loadB();
  storeB(0);
  if (wave == 0) tdm_wait();
  __syncthreads();

  const int KSTEPS = MDIM / 32;  // 24
  int rowA = waveM * 64 + (lane & 15);
  int colB = waveN * 32 + (lane & 15);
#pragma unroll 2
  for (int kbi = 0; kbi < KSTEPS; ++kbi) {
    int cur = kbi & 1, nxt = cur ^ 1;
    bool more = (kbi + 1) < KSTEPS;
    if (more) {
      if (wave == 0) issueA(nxt);
      aCur += 32;
      loadB();
    }

    BF16Frag aF[4], bF[2];
#pragma unroll
    for (int si = 0; si < 4; ++si) load_a_frag(aF[si], &sA[cur][rowA + si*16][0], lane);
#pragma unroll
    for (int sj = 0; sj < 2; ++sj) load_b_frag(bF[sj], &sB[cur][colB + sj*16][0], lane);
#pragma unroll
    for (int si = 0; si < 4; ++si)
#pragma unroll
      for (int sj = 0; sj < 2; ++sj)
        acc[si][sj].v = __builtin_amdgcn_wmma_f32_16x16x32_bf16(
            false, aF[si].v, false, bF[sj].v, (short)0, acc[si][sj].v, false, false);

    if (more) storeB(nxt);
    if (wave == 0) tdm_wait();
    __syncthreads();
  }

  int rowOff = (lane >> 4) * 8, nOff = lane & 15;
#pragma unroll
  for (int si = 0; si < 4; ++si)
#pragma unroll
    for (int sj = 0; sj < 2; ++sj)
#pragma unroll
      for (int i = 0; i < 8; ++i) {
        int gRow = mBase + waveM * 64 + si * 16 + rowOff + i;
        if (gRow < cnt)
          eout[(size_t)(rowStart + gRow) * DDIM + nBase + waveN*32 + sj*16 + nOff] =
              bfbits(acc[si][sj].f[i]);
      }
}

// ---------------- 6. unpermute + weighted combine (bf16 rows -> fp32 out) ----------------
__global__ __launch_bounds__(256) void combine_kernel(
    const float* __restrict__ rw, const int* __restrict__ pos_of,
    const unsigned short* __restrict__ eo, float* __restrict__ out)
{
  int t = blockIdx.x;
  int base = threadIdx.x * 8;   // 256 threads * 8 = 2048 = DDIM
  float a[8];
#pragma unroll
  for (int j = 0; j < 8; ++j) a[j] = 0.f;
#pragma unroll
  for (int k = 0; k < KTOP; ++k) {
    float w = rw[t*KTOP + k];
    int p = pos_of[t*KTOP + k];
    uint4 v = *(const uint4*)(eo + (size_t)p * DDIM + base);   // 8 bf16
    a[0] += w * bf2f_lo(v.x); a[1] += w * bf2f_hi(v.x);
    a[2] += w * bf2f_lo(v.y); a[3] += w * bf2f_hi(v.y);
    a[4] += w * bf2f_lo(v.z); a[5] += w * bf2f_hi(v.z);
    a[6] += w * bf2f_lo(v.w); a[7] += w * bf2f_hi(v.w);
  }
  float4* dst = (float4*)(out + (size_t)t * DDIM + base);
  float4 o0 = {a[0], a[1], a[2], a[3]};
  float4 o1 = {a[4], a[5], a[6], a[7]};
  dst[0] = o0; dst[1] = o1;
}

// ---------------- host launcher ----------------
extern "C" void kernel_launch(void* const* d_in, const int* in_sizes, int n_in,
                              void* d_out, int out_size, void* d_ws, size_t ws_size,
                              hipStream_t stream)
{
  (void)in_sizes; (void)n_in; (void)out_size; (void)ws_size;
  const float* x  = (const float*)d_in[0];   // [T, D]
  const float* gk = (const float*)d_in[1];   // [D, E]
  const float* wg = (const float*)d_in[2];   // [E, D, M]
  const float* wu = (const float*)d_in[3];   // [E, D, M]
  const float* wd = (const float*)d_in[4];   // [E, M, D]
  float* out = (float*)d_out;

  char* p = (char*)d_ws;
  auto alloc = [&](size_t bytes) -> char* {
    char* r = p; p += (bytes + 255) & ~(size_t)255; return r;
  };
  float*          rw      = (float*)alloc((size_t)TKROWS * 4);
  int*            sel     = (int*)  alloc((size_t)TKROWS * 4);
  int*            counts  = (int*)  alloc(NEXP * 4);
  int*            offsets = (int*)  alloc((NEXP + 1) * 4);
  int*            cursors = (int*)  alloc(NEXP * 4);
  int*            pos_of  = (int*)  alloc((size_t)TKROWS * 4);
  int*            row_tok = (int*)  alloc((size_t)TKROWS * 4);
  unsigned short* interbf = (unsigned short*)alloc((size_t)(TKROWS + G2_MT) * MDIM * 2);
  unsigned short* eout    = (unsigned short*)alloc((size_t)TKROWS * DDIM * 2);

  init_kernel  <<<1, 64, 0, stream>>>(counts, cursors);
  router_kernel<<<TOKENS / 8, 256, 0, stream>>>(x, gk, rw, sel, counts);
  scan_kernel  <<<1, 32, 0, stream>>>(counts, offsets);
  assign_kernel<<<TKROWS / 256, 256, 0, stream>>>(sel, offsets, cursors, pos_of, row_tok);

  dim3 g1(TKROWS / G1_MT, MDIM / G1_NT, NEXP);   // 256 x 12 x 32 (early-exit on empty tiles)
  gemm1_kernel<<<g1, 256, 0, stream>>>(x, wg, wu, row_tok, offsets, interbf);

  dim3 g2(TKROWS / G2_MT, DDIM / G2_NT, NEXP);   // 256 x 16 x 32
  gemm2_kernel<<<g2, 256, 0, stream>>>(interbf, wd, offsets, eout);

  combine_kernel<<<TOKENS, 256, 0, stream>>>(rw, pos_of, eout, out);
}
